// CNN_64570538328519
// MI455X (gfx1250) — matrix operand order repaired and checked by reading, not compiled, not run on hardware
//
#include <hip/hip_runtime.h>

typedef __attribute__((ext_vector_type(16))) _Float16 v16h;
typedef __attribute__((ext_vector_type(8)))  float    v8f;

#define W_TOT  3044
#define BN_EPS 1e-5f

__constant__ int CHR_OFF[24] = {
    0, 249, 492, 691, 882, 1064, 1235, 1395, 1541, 1680, 1814, 1950,
    2084, 2199, 2307, 2409, 2500, 2584, 2665, 2724, 2789, 2836, 2887, 3044};

// ---------------------------------------------------------------------------
// Kernel 1: one workgroup (8 waves) per sample. Streams x2 once, then both
// conv layers run as implicit-GEMM WMMA (f16 in / f32 acc) out of LDS.
// Emits 224-wide f16 feature row (25 x1 | 192 conv | 5 x2_s | 1 gini | pad).
// K' remap: conv rows padded 45->48 so per-lane 8-element A runs are
// contiguous in LDS; K-padding carries zero weights (A pads never NaN).
// ---------------------------------------------------------------------------
__global__ __launch_bounds__(256) void feat_kernel(
    const float* __restrict__ x2, const int* __restrict__ coord,
    const float* __restrict__ c1w, const float* __restrict__ c1b,
    const float* __restrict__ b1g, const float* __restrict__ b1b,
    const float* __restrict__ b1m, const float* __restrict__ b1v,
    const float* __restrict__ c2w, const float* __restrict__ c2b,
    const float* __restrict__ b2g, const float* __restrict__ b2b,
    const float* __restrict__ b2m, const float* __restrict__ b2v,
    _Float16* __restrict__ featH)
{
    __shared__ _Float16 xb[6][3048];     // rows 0-4 binarized (+2 pad cols); row 5 zeros
    __shared__ _Float16 w1t[16][256];    // conv1 weights, [cout][K'=r*48+k], zero-padded
    __shared__ _Float16 w2t[16][384];    // conv2 weights, [cout][K'=c1*48+k], zero-padded
    __shared__ _Float16 p1h[8][176];     // pooled conv1 out, cols 137..175 zeroed
    __shared__ float    chrsum[23];
    __shared__ float    sS[5];
    __shared__ float    feat[224];
    __shared__ float    totS;

    const int   tid   = threadIdx.x;
    const int   b     = blockIdx.x;
    const float* xbase = x2 + (size_t)b * (5 * W_TOT);

    // ---- init LDS ----
    if (tid < 23) chrsum[tid] = 0.f;
    if (tid >= 32 && tid < 37) sS[tid - 32] = 0.f;
    if (tid >= 64 && tid < 84) {                  // xb edge pad columns
        int r = (tid - 64) >> 2, q = (tid - 64) & 3;
        int c = (q < 2) ? q : (3044 + q);         // 0,1,3046,3047
        xb[r][c] = (_Float16)0.f;
    }
    if (tid == 96) feat[223] = 0.f;
    for (int i = tid; i < 3048; i += 256) xb[5][i] = (_Float16)0.f;   // K'-pad row
    for (int i = tid; i < 312; i += 256) {                            // p1 col pad
        int c = i / 39, col = 137 + i % 39;
        p1h[c][col] = (_Float16)0.f;
    }
    for (int i = tid; i < 16 * 256; i += 256) {   // conv1 weights -> [cout][K']
        int c = i >> 8, Kp = i & 255;
        int r = Kp / 48, k = Kp % 48;
        float w = (c < 8 && r < 5 && k < 45) ? c1w[c * 225 + r * 45 + k] : 0.f;
        w1t[c][Kp] = (_Float16)w;
    }
    for (int i = tid; i < 16 * 384; i += 256) {   // conv2 weights -> [cout][K']
        int c = i / 384, Kp = i % 384;
        int c1 = Kp / 48, k = Kp % 48;
        float w = (k < 45) ? c2w[c * 360 + c1 * 45 + k] : 0.f;
        w2t[c][Kp] = (_Float16)w;
    }
    __syncthreads();

    // ---- phase 1: single float4 pass: binarize, row sums, segment sums ----
    float sloc[5] = {0.f, 0.f, 0.f, 0.f, 0.f};
    const float4* x4 = (const float4*)xbase;
    for (int i4 = tid; i4 < 5 * (W_TOT / 4); i4 += 256) {
        __builtin_prefetch((const void*)(x4 + i4 + 1024), 0, 0);   // global_prefetch_b8
        float4 v = x4[i4];
        int r  = i4 / 761;
        int c0 = (i4 - r * 761) * 4;
        float vv[4] = {v.x, v.y, v.z, v.w};
#pragma unroll
        for (int q = 0; q < 4; ++q) {
            float bv = (vv[q] > 0.f) ? 1.f : 0.f;
            xb[r][c0 + q + 2] = (_Float16)bv;
            sloc[r] += bv;
        }
        if (r == 2) {
            int s = 0;
            while (CHR_OFF[s + 1] <= c0) s++;
#pragma unroll
            for (int q = 0; q < 4; ++q) {
                while (CHR_OFF[s + 1] <= c0 + q) s++;
                atomicAdd(&chrsum[s], vv[q]);
            }
        }
    }
#pragma unroll
    for (int r = 0; r < 5; ++r) atomicAdd(&sS[r], sloc[r]);
    __syncthreads();

    // ---- phase 2a: x1 slice + gini (runs alongside conv1 on wave 0) ----
    if (tid < 25) {
        int r = tid / 5, c = tid % 5;
        int st = coord[b] - 2;
        float v = xbase[r * W_TOT + st + c];
        feat[tid] = log2f(v + 1.f);
    }
    if (tid == 25) {
        float xs[23];
        for (int k = 0; k < 23; ++k)
            xs[k] = chrsum[k] / (float)(CHR_OFF[k + 1] - CHR_OFF[k]);
        for (int i = 1; i < 23; ++i) {
            float key = xs[i]; int j = i - 1;
            while (j >= 0 && xs[j] > key) { xs[j + 1] = xs[j]; --j; }
            xs[j + 1] = key;
        }
        float num = 0.f, den = 0.f;
        for (int i = 0; i < 23; ++i) { num += (float)(2 * (i + 1) - 24) * xs[i]; den += xs[i]; }
        feat[222] = num / (23.f * den);
        totS = sS[0] + sS[1] + sS[2] + sS[3] + sS[4];
    }

    // ---- conv1 as WMMA implicit GEMM: M=288(18 tiles), N=16, K=256 ----
    const int lane  = tid & 31;
    const int wave  = tid >> 5;
    const int mrow  = lane & 15;
    const int khalf = lane >> 4;
    const int kbase = khalf * 8;        // A lane-group K offset
    const int koff  = khalf * 8;        // B lane-group K offset
    const int cN    = mrow;             // output-channel column
    const int cc1   = cN & 7;           // safe param index for conv1

    for (int mt = wave; mt < 18; mt += 8) {
        int jrow = mt * 16 + mrow;      // conv1 output position (A row)
        v8f acc = {};
#pragma unroll
        for (int ks = 0; ks < 8; ++ks) {
            int K1 = ks * 32 + kbase;
            int K2 = K1 + 16;
            const _Float16* a1 = &xb[K1 / 48][jrow * 11 + (K1 % 48)];
            const _Float16* a2 = &xb[K2 / 48][jrow * 11 + (K2 % 48)];
            v16h a;
#pragma unroll
            for (int e = 0; e < 8; ++e) { a[e] = a1[e]; a[8 + e] = a2[e]; }
            const _Float16* bp = &w1t[cN][ks * 32 + koff];
            v16h bm;
#pragma unroll
            for (int e = 0; e < 8; ++e) { bm[e] = bp[e]; bm[8 + e] = bp[16 + e]; }
            acc = __builtin_amdgcn_wmma_f32_16x16x32_f16(
                false, a, false, bm, (short)0, acc, false, false);
        }
        // epilogue: bias + bn + relu + maxpool(1,2); C: M = i + 8*khalf
        float scale = b1g[cc1] * rsqrtf(b1v[cc1] + BN_EPS);
        float bias = c1b[cc1], bmu = b1m[cc1], bbe = b1b[cc1];
#pragma unroll
        for (int i = 0; i < 8; i += 2) {
            float z0 = (acc[i]     + bias - bmu) * scale + bbe;
            float z1 = (acc[i + 1] + bias - bmu) * scale + bbe;
            float p  = fmaxf(fmaxf(z0, 0.f), fmaxf(z1, 0.f));
            int jp = mt * 8 + khalf * 4 + (i >> 1);
            if (cN < 8 && jp < 137) p1h[cN][jp] = (_Float16)p;
        }
    }
    __syncthreads();

    // ---- phase 2b: x2_s ----
    if (tid >= 224 && tid < 229) {
        int r = tid - 224;
        feat[217 + r] = sS[r] / fmaxf(totS, 1e-12f);
    }

    // ---- conv2 as WMMA implicit GEMM: M=32(2 tiles), N=16, K=384 ----
    if (wave < 2) {
        int mt   = wave;
        int jrow = mt * 16 + mrow;      // conv2 output position
        v8f acc = {};
#pragma unroll
        for (int ks = 0; ks < 12; ++ks) {
            int K1 = ks * 32 + kbase;
            int K2 = K1 + 16;
            const _Float16* a1 = &p1h[K1 / 48][jrow * 4 + (K1 % 48)];
            const _Float16* a2 = &p1h[K2 / 48][jrow * 4 + (K2 % 48)];
            v16h a;
#pragma unroll
            for (int e = 0; e < 8; ++e) { a[e] = a1[e]; a[8 + e] = a2[e]; }
            const _Float16* bp = &w2t[cN][ks * 32 + koff];
            v16h bm;
#pragma unroll
            for (int e = 0; e < 8; ++e) { bm[e] = bp[e]; bm[8 + e] = bp[16 + e]; }
            acc = __builtin_amdgcn_wmma_f32_16x16x32_f16(
                false, a, false, bm, (short)0, acc, false, false);
        }
        float scale = b2g[cN] * rsqrtf(b2v[cN] + BN_EPS);
        float bias = c2b[cN], bmu = b2m[cN], bbe = b2b[cN];
#pragma unroll
        for (int i = 0; i < 8; i += 2) {
            float z0 = (acc[i]     + bias - bmu) * scale + bbe;
            float z1 = (acc[i + 1] + bias - bmu) * scale + bbe;
            float p  = fmaxf(fmaxf(z0, 0.f), fmaxf(z1, 0.f));
            int jp = mt * 8 + khalf * 4 + (i >> 1);
            if (jp < 12) feat[25 + cN * 12 + jp] = p;
        }
    }
    __syncthreads();

    // ---- emit f16 feature row ----
    if (tid < 224) featH[(size_t)b * 224 + tid] = (_Float16)feat[tid];
}

// ---------------------------------------------------------------------------
// Kernel 2: WMMA GEMM  z = feat(8192x224) @ fc1_w^T(224x64), fused
// bias + batchnorm + exact GELU + fc2 dot-product epilogue.
// ---------------------------------------------------------------------------
__global__ __launch_bounds__(256) void fc_kernel(
    const _Float16* __restrict__ featH,
    const float* __restrict__ w1,  const float* __restrict__ b1,
    const float* __restrict__ gg,  const float* __restrict__ gb,
    const float* __restrict__ gm,  const float* __restrict__ gv,
    const float* __restrict__ w2,  const float* __restrict__ b2,
    float* __restrict__ out)
{
    __shared__ _Float16 WH[64 * 224];   // fc1_w as f16, K-padded to 224
    __shared__ float    rowsum[128];

    const int tid = threadIdx.x;
    for (int i = tid; i < 64 * 224; i += 256) {
        int n = i / 224, k = i % 224;
        WH[i] = (k < 223) ? (_Float16)w1[n * 223 + k] : (_Float16)0.f;
    }
    if (tid < 128) rowsum[tid] = 0.f;
    __syncthreads();

    const int lane  = tid & 31;
    const int wave  = tid >> 5;
    const int mrow  = lane & 15;
    const int khalf = lane >> 4;
    const int m0    = blockIdx.x * 128 + wave * 16;

    v8f acc[4] = {};
    for (int k0 = 0; k0 < 224; k0 += 32) {
        const _Float16* ap = featH + (size_t)(m0 + mrow) * 224 + k0 + khalf * 8;
        v16h a;
#pragma unroll
        for (int e = 0; e < 8; ++e) a[e] = ap[e];
#pragma unroll
        for (int e = 0; e < 8; ++e) a[8 + e] = ap[16 + e];
#pragma unroll
        for (int nt = 0; nt < 4; ++nt) {
            const _Float16* bp = &WH[(nt * 16 + mrow) * 224 + k0 + khalf * 8];
            v16h bm;
#pragma unroll
            for (int e = 0; e < 8; ++e) { bm[e] = bp[e]; bm[8 + e] = bp[16 + e]; }
            acc[nt] = __builtin_amdgcn_wmma_f32_16x16x32_f16(
                false, a, false, bm, (short)0, acc[nt], false, false);
        }
    }

#pragma unroll
    for (int nt = 0; nt < 4; ++nt) {
        int   n     = nt * 16 + mrow;
        float scale = gg[n] * rsqrtf(gv[n] + BN_EPS);
        float wc    = w2[n];
#pragma unroll
        for (int i = 0; i < 8; ++i) {
            int   lr = wave * 16 + i + khalf * 8;
            float z  = acc[nt][i] + b1[n];
            z = (z - gm[n]) * scale + gb[n];
            float gl = 0.5f * z * (1.f + erff(z * 0.70710678118654752f));
            atomicAdd(&rowsum[lr], gl * wc);
        }
    }
    __syncthreads();
    if (tid < 128) out[blockIdx.x * 128 + tid] = rowsum[tid] + b2[0];
}

// ---------------------------------------------------------------------------
extern "C" void kernel_launch(void* const* d_in, const int* in_sizes, int n_in,
                              void* d_out, int out_size, void* d_ws, size_t ws_size,
                              hipStream_t stream)
{
    const float* x2    = (const float*)d_in[0];
    const int*   coord = (const int*)  d_in[1];
    const float* c1w = (const float*)d_in[2];
    const float* c1b = (const float*)d_in[3];
    const float* b1g = (const float*)d_in[4];
    const float* b1b = (const float*)d_in[5];
    const float* b1m = (const float*)d_in[6];
    const float* b1v = (const float*)d_in[7];
    const float* c2w = (const float*)d_in[8];
    const float* c2b = (const float*)d_in[9];
    const float* b2g = (const float*)d_in[10];
    const float* b2b = (const float*)d_in[11];
    const float* b2m = (const float*)d_in[12];
    const float* b2v = (const float*)d_in[13];
    const float* f1w = (const float*)d_in[14];
    const float* f1b = (const float*)d_in[15];
    const float* bfg = (const float*)d_in[16];
    const float* bfb = (const float*)d_in[17];
    const float* bfm = (const float*)d_in[18];
    const float* bfv = (const float*)d_in[19];
    const float* f2w = (const float*)d_in[20];
    const float* f2b = (const float*)d_in[21];

    const int B = in_sizes[1];                  // 8192 samples
    _Float16* featH = (_Float16*)d_ws;          // B x 224 f16 features

    feat_kernel<<<B, 256, 0, stream>>>(x2, coord,
                                       c1w, c1b, b1g, b1b, b1m, b1v,
                                       c2w, c2b, b2g, b2b, b2m, b2v, featH);
    fc_kernel<<<B / 128, 256, 0, stream>>>(featH, f1w, f1b,
                                           bfg, bfb, bfm, bfv,
                                           f2w, f2b, (float*)d_out);
}
